// Node_Policy_13477607375252
// MI455X (gfx1250) — matrix operand order repaired and checked by reading, not compiled, not run on hardware
//
#include <hip/hip_runtime.h>
#include <hip/hip_bf16.h>
#include <math.h>

typedef __attribute__((ext_vector_type(16))) __bf16 v16bf;
typedef __attribute__((ext_vector_type(8)))  float  v8f;
typedef __attribute__((ext_vector_type(4)))  int    v4i;
typedef unsigned short u16;

#define N_NODES 8192
#define E_EDGES 16384
#define FDIM    16
#define H1D     128
#define H2D     64
#define A_ACT   4096
#define PDIM    32
#define N_STEPS 6
#define KAUG    16512   // 128*128 outer-product K + 128 bias rows
#define KBLKS   516     // KAUG / 32
#define NTILES  8       // H1D / 16
#define CH      4       // k-values staged per LDS chunk

#if defined(__has_builtin)
#if __has_builtin(__builtin_amdgcn_global_load_async_to_lds_b128) && \
    __has_builtin(__builtin_amdgcn_s_wait_asynccnt)
#define USE_ASYNC_LDS 1
#endif
#endif

#ifdef USE_ASYNC_LDS
typedef __attribute__((address_space(1))) v4i* gas_v4i_p;   // global (AS1) int4*
typedef __attribute__((address_space(3))) v4i* lds_v4i_p;   // LDS (AS3) int4*
#endif

// ---------- small helpers ----------
__device__ __forceinline__ u16 f2bf(float v) {
    __bf16 b = (__bf16)v;
    return __builtin_bit_cast(u16, b);
}
__device__ __forceinline__ float bf2f(u16 u) {
    __bf16 b = __builtin_bit_cast(__bf16, u);
    return (float)b;
}
// 16-byte global -> LDS copy (async path when available)
__device__ __forceinline__ void stage16B(const u16* g, u16* l) {
#ifdef USE_ASYNC_LDS
    __builtin_amdgcn_global_load_async_to_lds_b128(
        (gas_v4i_p)g, (lds_v4i_p)l, 0, 0);
#else
    *(uint4*)l = *(const uint4*)g;
#endif
}
__device__ __forceinline__ void async_wait_all() {
#ifdef USE_ASYNC_LDS
    __builtin_amdgcn_s_wait_asynccnt(0);
#endif
}

// ---------- column mean / 1/(std+eps) over rows ----------
__global__ __launch_bounds__(256)
void colstats_kernel(const float* __restrict__ x, int rows, int cols,
                     float ddof, float eps,
                     float* __restrict__ mean_out, float* __restrict__ rstd_out) {
    int c = blockIdx.x;
    __shared__ float ssum[256], ssq[256];
    float s = 0.f, q = 0.f;
    for (int r = threadIdx.x; r < rows; r += 256) {
        float v = x[(size_t)r * cols + c];
        s += v; q += v * v;
    }
    ssum[threadIdx.x] = s; ssq[threadIdx.x] = q;
    __syncthreads();
    for (int w = 128; w > 0; w >>= 1) {
        if (threadIdx.x < w) {
            ssum[threadIdx.x] += ssum[threadIdx.x + w];
            ssq [threadIdx.x] += ssq [threadIdx.x + w];
        }
        __syncthreads();
    }
    if (threadIdx.x == 0) {
        float rn  = (float)rows;
        float m   = ssum[0] / rn;
        float var = ssq[0] / rn - m * m;
        var *= rn / (rn - ddof);          // ddof=0 (population) or 1 (unbiased)
        if (var < 0.f) var = 0.f;
        mean_out[c] = m;
        rstd_out[c] = 1.0f / (sqrtf(var) + eps);
    }
}

// ---------- h0 = relu(norm(feat) @ Wp + bp) ----------
__global__ __launch_bounds__(128)
void h0_kernel(const float* __restrict__ feat, const float* __restrict__ Wp,
               const float* __restrict__ bp, const float* __restrict__ fm,
               const float* __restrict__ fr, float* __restrict__ hidden) {
    int n = blockIdx.x, j = threadIdx.x;
    float acc = bp[j];
#pragma unroll
    for (int f = 0; f < FDIM; ++f) {
        float nf = (feat[n * FDIM + f] - fm[f]) * fr[f];
        acc += nf * Wp[f * H1D + j];
    }
    hidden[(size_t)n * H1D + j] = fmaxf(acc, 0.f);
}

// ---------- a_e = relu(norm(edge_feat_e) * We1 + be1), stored bf16 ----------
__global__ __launch_bounds__(128)
void edge_a_kernel(const float* __restrict__ ew, const float* __restrict__ We1,
                   const float* __restrict__ be1, const float* __restrict__ ewm,
                   const float* __restrict__ ewr, u16* __restrict__ a_out) {
    int e = blockIdx.x, k = threadIdx.x;
    float ne = (ew[e] - ewm[0]) * ewr[0];
    float v  = fmaxf(ne * We1[k] + be1[k], 0.f);
    a_out[(size_t)e * H1D + k] = f2bf(v);
}

// ---------- pack augmented We2 (16512 x 128) into WMMA-B lane layout, bf16 ----------
__global__ __launch_bounds__(256)
void pack_wb_kernel(const float* __restrict__ We2, const float* __restrict__ be2,
                    u16* __restrict__ Wb) {
    int t = blockIdx.x * 256 + threadIdx.x;
    const int TOT = NTILES * KBLKS * 32 * 16;
    if (t >= TOT) return;
    int p    = t & 15;
    int lane = (t >> 4) & 31;
    int kb   = (t >> 9) % KBLKS;
    int nt   = t / (512 * KBLKS);
    int row  = kb * 32 + p + ((lane >> 4) << 4);
    int col  = nt * 16 + (lane & 15);
    float v;
    if (row < 16384) v = We2[(size_t)row * H1D + col];        // flat (128,16384) == (16384,128) here
    else             v = be2[(size_t)(row - 16384) * H1D + col];
    Wb[t] = f2bf(v);
}

// ---------- fused msg GEMM + scatter-add:  agg[dst] += (a (x) h_src | h_src) @ Waug ----------
// 4 waves / block walk the identical kb sequence -> B tiles staged once per block into
// double-buffered LDS (async global->LDS when available); 4x cut in L2 traffic.
__global__ __launch_bounds__(128)
void msg_gemm_kernel(const float* __restrict__ hidden, const u16* __restrict__ a_bf,
                     const u16* __restrict__ Wb, const int* __restrict__ src,
                     const int* __restrict__ dst, float* __restrict__ agg) {
    const int tid  = threadIdx.x;
    const int wave = tid >> 5;
    const int lane = tid & 31;
    __shared__ u16 sh_h[4][16 * H1D];
    __shared__ u16 sh_a[4][16 * H1D];
    __shared__ int sh_dst[4][16];
    __shared__ u16 sB[2][CH * NTILES * 512];   // 2 x 32KB B staging

    {
        int e0 = (blockIdx.x * 4 + wave) * 16;
        for (int t = lane; t < 16 * H1D; t += 32) {
            int m = t >> 7, c = t & 127;
            int e = e0 + m;
            sh_h[wave][t] = f2bf(hidden[(size_t)src[e] * H1D + c]);
            sh_a[wave][t] = a_bf[(size_t)e * H1D + c];
        }
        if (lane < 16) sh_dst[wave][lane] = dst[e0 + lane];
    }
    __syncthreads();

    v8f acc[NTILES];
#pragma unroll
    for (int i = 0; i < NTILES; ++i) acc[i] = v8f{0.f,0.f,0.f,0.f,0.f,0.f,0.f,0.f};

    const int half = lane >> 4;    // K sub-block for A layout
    const int mrow = lane & 15;    // A-matrix row handled by this lane
    const u16* sha = &sh_a[wave][mrow * H1D];
    const u16* shh = &sh_h[wave][mrow * H1D];

    // stage B for k = kc..kc+CH-1 (kb = 4k+sub): 2048 x 16B pieces over 128 threads
    auto stage = [&](int kc, int sub, int buf) {
#pragma unroll
        for (int i = 0; i < 16; ++i) {
            int pi     = tid + 128 * i;
            int region = pi >> 6;            // c*8 + nt  (1KB regions)
            int within = pi & 63;            // 16B piece inside region
            int c  = region >> 3;
            int nt = region & 7;
            int kb = (kc + c) * 4 + sub;
            const u16* g = Wb + ((size_t)nt * KBLKS + kb) * 512 + within * 8;
            stage16B(g, &sB[buf][region * 512 + within * 8]);
        }
    };
    auto wmma8lds = [&](const v16bf& av, int buf, int c) {
#pragma unroll
        for (int nt = 0; nt < NTILES; ++nt) {
            const v16bf bv = *(const v16bf*)&sB[buf][(c * NTILES + nt) * 512 + lane * 16];
            acc[nt] = __builtin_amdgcn_wmma_f32_16x16x32_bf16(
                false, av, false, bv, (short)0, acc[nt], false, false);
        }
    };

#pragma unroll 1
    for (int sub = 0; sub < 4; ++sub) {
        // 16 h-values for this sub, fixed across all 128 k iterations
        float hvf[16];
#pragma unroll
        for (int p = 0; p < 16; ++p) {
            int koff = p + (half ? 16 : 0);
            hvf[p] = bf2f(shh[sub * 32 + koff]);
        }
        stage(0, sub, 0);
        async_wait_all();
        __syncthreads();
#pragma unroll 1
        for (int kc = 0; kc < 128; kc += CH) {
            int buf = (kc / CH) & 1;
            if (kc + CH < 128) stage(kc + CH, sub, buf ^ 1);   // prefetch next chunk
#pragma unroll
            for (int c = 0; c < CH; ++c) {
                float as = bf2f(sha[kc + c]);
                v16bf av;
#pragma unroll
                for (int p = 0; p < 16; ++p) av[p] = (__bf16)(as * hvf[p]);
                wmma8lds(av, buf, c);
            }
            async_wait_all();      // next chunk staged
            __syncthreads();       // all waves done with 'buf' before it is reused
        }
    }

    // ---- bias tail: kb in [512, 516), A element = h[i], B direct from global
#pragma unroll 1
    for (int kb = 512; kb < 516; ++kb) {
        int ibase = (kb - 512) * 32;
        v16bf bA[NTILES];
#pragma unroll
        for (int nt = 0; nt < NTILES; ++nt)
            bA[nt] = *(const v16bf*)(Wb + (((size_t)nt * KBLKS + kb) * 32 + lane) * 16);
        v16bf av;
#pragma unroll
        for (int p = 0; p < 16; ++p) {
            int koff = p + (half ? 16 : 0);
            av[p] = __builtin_bit_cast(__bf16, shh[ibase + koff]);
        }
#pragma unroll
        for (int nt = 0; nt < NTILES; ++nt)
            acc[nt] = __builtin_amdgcn_wmma_f32_16x16x32_bf16(
                false, av, false, bA[nt], (short)0, acc[nt], false, false);
    }

    // C layout: element r of acc -> row M = r + 8*half, col N = (lane&15) + nt*16
#pragma unroll
    for (int nt = 0; nt < NTILES; ++nt) {
        int col = nt * 16 + (lane & 15);
#pragma unroll
        for (int r = 0; r < 8; ++r) {
            int m = r + 8 * half;
            int d = sh_dst[wave][m];
            atomicAdd(&agg[(size_t)d * H1D + col], acc[nt][r]);
        }
    }
}

// ---------- GRU cell, 8 nodes per block (8x weight reuse) ----------
#define GRU_NB 8
__global__ __launch_bounds__(128)
void gru_kernel(const float* __restrict__ agg, const float* __restrict__ conv_b,
                const float* __restrict__ Wi, const float* __restrict__ Wh,
                const float* __restrict__ bi, const float* __restrict__ bh,
                float* __restrict__ hidden) {
    int n0 = blockIdx.x * GRU_NB, j = threadIdx.x;
    __shared__ float sx[GRU_NB][H1D], sh[GRU_NB][H1D];
    float hv[GRU_NB];
#pragma unroll
    for (int q = 0; q < GRU_NB; ++q) {
        size_t o = (size_t)(n0 + q) * H1D + j;
        hv[q] = hidden[o];
        sx[q][j] = fmaxf(agg[o] + conv_b[j], 0.f);
        sh[q][j] = hv[q];
    }
    __syncthreads();
    float gr[GRU_NB], gz[GRU_NB], gn[GRU_NB], ar[GRU_NB], az[GRU_NB], an[GRU_NB];
#pragma unroll
    for (int q = 0; q < GRU_NB; ++q) {
        gr[q] = bi[j];  gz[q] = bi[H1D + j];  gn[q] = bi[2 * H1D + j];
        ar[q] = bh[j];  az[q] = bh[H1D + j];  an[q] = bh[2 * H1D + j];
    }
    for (int i = 0; i < H1D; ++i) {
        float wir = Wi[i * 384 + j], wiz = Wi[i * 384 + H1D + j], win = Wi[i * 384 + 2 * H1D + j];
        float whr = Wh[i * 384 + j], whz = Wh[i * 384 + H1D + j], whn = Wh[i * 384 + 2 * H1D + j];
#pragma unroll
        for (int q = 0; q < GRU_NB; ++q) {
            float xv = sx[q][i], hh = sh[q][i];
            gr[q] += xv * wir;  gz[q] += xv * wiz;  gn[q] += xv * win;
            ar[q] += hh * whr;  az[q] += hh * whz;  an[q] += hh * whn;
        }
    }
#pragma unroll
    for (int q = 0; q < GRU_NB; ++q) {
        float r  = 1.f / (1.f + expf(-(gr[q] + ar[q])));
        float z  = 1.f / (1.f + expf(-(gz[q] + az[q])));
        float nn = tanhf(gn[q] + r * an[q]);
        hidden[(size_t)(n0 + q) * H1D + j] = (1.f - z) * nn + z * hv[q];
    }
}

// ---------- elementwise leaky-relu (in place) ----------
__global__ __launch_bounds__(256)
void lrelu_kernel(float* __restrict__ x, int n) {
    int t = blockIdx.x * 256 + threadIdx.x;
    if (t < n) { float v = x[t]; x[t] = v > 0.f ? v : 0.1f * v; }
}

// ---------- level embedding sum over P gathered rows ----------
__global__ __launch_bounds__(128)
void level_sum_kernel(const float* __restrict__ mpnn, const int* __restrict__ idx,
                      float* __restrict__ s_out) {
    int n = blockIdx.x, j = threadIdx.x;
    __shared__ int si[PDIM];
    if (j < PDIM) si[j] = idx[n * PDIM + j];
    __syncthreads();
    float acc = 0.f;
#pragma unroll 4
    for (int p = 0; p < PDIM; ++p) acc += mpnn[(size_t)si[p] * H1D + j];
    s_out[(size_t)n * H1D + j] = acc;
}

// ---------- normalize in place with precomputed stats ----------
__global__ __launch_bounds__(256)
void norm_inplace_kernel(float* __restrict__ x, const float* __restrict__ mean,
                         const float* __restrict__ rstd, int n, int cols) {
    int t = blockIdx.x * 256 + threadIdx.x;
    if (t < n) { int c = t % cols; x[t] = (x[t] - mean[c]) * rstd[c]; }
}

// ---------- row gather ----------
__global__ __launch_bounds__(256)
void gather_kernel(const float* __restrict__ src_, const int* __restrict__ idx,
                   float* __restrict__ out_, int n, int cols) {
    int t = blockIdx.x * 256 + threadIdx.x;
    if (t < n) {
        int a = t / cols, c = t - a * cols;
        out_[t] = src_[(size_t)idx[a] * cols + c];
    }
}

// ---------- assemble rep = [lrelu(norm(las)@W1+b1) | mpnn[legal] | norm(gb) | norm(gt)] ----------
__global__ __launch_bounds__(512)
void build_rep_kernel(const float* __restrict__ lasg, const float* __restrict__ lasm,
                      const float* __restrict__ lasr, const float* __restrict__ W1,
                      const float* __restrict__ b1, const float* __restrict__ mpnn,
                      const int* __restrict__ legal,
                      const float* __restrict__ gb, const float* __restrict__ gbm,
                      const float* __restrict__ gbr,
                      const float* __restrict__ gt, const float* __restrict__ gtm,
                      const float* __restrict__ gtr, float* __restrict__ rep) {
    int a = blockIdx.x, j = threadIdx.x;
    float v;
    if (j < 128) {
        float acc = b1[j];
#pragma unroll
        for (int f = 0; f < FDIM; ++f)
            acc += (lasg[a * FDIM + f] - lasm[f]) * lasr[f] * W1[f * H1D + j];
        v = acc > 0.f ? acc : 0.1f * acc;
    } else if (j < 256) {
        v = mpnn[(size_t)legal[a] * H1D + (j - 128)];
    } else if (j < 384) {
        int c = j - 256; v = (gb[(size_t)a * H1D + c] - gbm[c]) * gbr[c];
    } else {
        int c = j - 384; v = (gt[(size_t)a * H1D + c] - gtm[c]) * gtr[c];
    }
    rep[(size_t)a * 512 + j] = v;
}

// ---------- head: out = lrelu(rep @ W2 + b2) @ W3 + b3 ----------
__global__ __launch_bounds__(64)
void head_kernel(const float* __restrict__ rep, const float* __restrict__ W2,
                 const float* __restrict__ b2, const float* __restrict__ W3,
                 const float* __restrict__ b3, float* __restrict__ out) {
    int a = blockIdx.x, j = threadIdx.x;
    __shared__ float srep[512];
    __shared__ float sred[64];
    for (int t = j; t < 512; t += 64) srep[t] = rep[(size_t)a * 512 + t];
    __syncthreads();
    float acc = b2[j];
    for (int i = 0; i < 512; ++i) acc += srep[i] * W2[i * H2D + j];
    acc = acc > 0.f ? acc : 0.1f * acc;
    sred[j] = acc * W3[j];
    __syncthreads();
    for (int w = 32; w > 0; w >>= 1) {
        if (j < w) sred[j] += sred[j + w];
        __syncthreads();
    }
    if (j == 0) out[a] = sred[0] + b3[0];
}

// =====================================================================
static inline size_t alignUp(size_t x, size_t a) { return (x + a - 1) & ~(a - 1); }

extern "C" void kernel_launch(void* const* d_in, const int* in_sizes, int n_in,
                              void* d_out, int out_size, void* d_ws, size_t ws_size,
                              hipStream_t stream) {
    (void)in_sizes; (void)n_in; (void)out_size; (void)ws_size;
    const float* feat      = (const float*)d_in[0];
    const float* edge_feat = (const float*)d_in[1];
    const int*   src       = (const int*)d_in[2];
    const int*   dst       = (const int*)d_in[3];
    const int*   legal     = (const int*)d_in[4];
    const int*   b_idx     = (const int*)d_in[5];
    const int*   t_idx     = (const int*)d_in[6];
    /* d_in[7] = curr_step (0 path) */
    const float* Wp   = (const float*)d_in[8];
    const float* bp   = (const float*)d_in[9];
    const float* We1  = (const float*)d_in[10];
    const float* be1  = (const float*)d_in[11];
    const float* We2  = (const float*)d_in[12];
    const float* be2  = (const float*)d_in[13];
    const float* conv_b = (const float*)d_in[14];
    const float* Wi   = (const float*)d_in[15];
    const float* Wh   = (const float*)d_in[16];
    const float* bi   = (const float*)d_in[17];
    const float* bh   = (const float*)d_in[18];
    const float* W1   = (const float*)d_in[19];
    const float* b1   = (const float*)d_in[20];
    const float* W2   = (const float*)d_in[21];
    const float* b2   = (const float*)d_in[22];
    const float* W3   = (const float*)d_in[23];
    const float* b3   = (const float*)d_in[24];
    float* out = (float*)d_out;

    // ---- workspace layout (floats), 256B aligned regions ----
    float* wsf = (float*)d_ws;
    size_t off = 0;
    float* stats  = wsf + off; off = alignUp(off + 2048, 64);
    float* hidden = wsf + off; off = alignUp(off + (size_t)N_NODES * H1D, 64);
    float* agg    = wsf + off; off = alignUp(off + (size_t)N_NODES * H1D, 64);  // reused as s_b
    float* s_t    = wsf + off; off = alignUp(off + (size_t)N_NODES * H1D, 64);
    float* gb     = wsf + off; off = alignUp(off + (size_t)A_ACT * H1D, 64);
    float* gt     = wsf + off; off = alignUp(off + (size_t)A_ACT * H1D, 64);
    float* lasg   = wsf + off; off = alignUp(off + (size_t)A_ACT * FDIM, 64);
    float* rep    = wsf + off; off = alignUp(off + (size_t)A_ACT * 512, 64);
    u16* a_bf = (u16*)(wsf + off); off = alignUp(off + (size_t)E_EDGES * H1D / 2, 64);
    u16* Wb   = (u16*)(wsf + off); off = alignUp(off + (size_t)NTILES * KBLKS * 32 * 16 / 2, 64);

    // stats slots
    float* fm  = stats +   0; float* fr  = stats +  16;   // feat cols (16)
    float* ewm = stats +  32; float* ewr = stats +  33;   // edge col (1)
    float* bM  = stats +  64; float* bR  = stats + 192;   // s_b cols (128), ddof=1
    float* tM  = stats + 320; float* tR  = stats + 448;   // s_t cols (128), ddof=1
    float* lm  = stats + 576; float* lr  = stats + 592;   // las cols (16)
    float* gbm = stats + 608; float* gbr = stats + 736;   // gb cols (128)
    float* gtm = stats + 864; float* gtr = stats + 992;   // gt cols (128)

    // 1) input normalization stats
    colstats_kernel<<<FDIM, 256, 0, stream>>>(feat, N_NODES, FDIM, 0.f, 1e-6f, fm, fr);
    colstats_kernel<<<1,    256, 0, stream>>>(edge_feat, E_EDGES, 1, 0.f, 1e-6f, ewm, ewr);

    // 2) node projection + edge gate vectors + packed bf16 weight
    h0_kernel<<<N_NODES, 128, 0, stream>>>(feat, Wp, bp, fm, fr, hidden);
    edge_a_kernel<<<E_EDGES, 128, 0, stream>>>(edge_feat, We1, be1, ewm, ewr, a_bf);
    {
        int tot = NTILES * KBLKS * 32 * 16;
        pack_wb_kernel<<<(tot + 255) / 256, 256, 0, stream>>>(We2, be2, Wb);
    }

    // 3) message-passing steps: fused WMMA GEMM + atomic scatter, then GRU
    for (int step = 0; step < N_STEPS; ++step) {
        (void)hipMemsetAsync(agg, 0, (size_t)N_NODES * H1D * sizeof(float), stream);
        msg_gemm_kernel<<<E_EDGES / 16 / 4, 128, 0, stream>>>(hidden, a_bf, Wb, src, dst, agg);
        gru_kernel<<<N_NODES / GRU_NB, 128, 0, stream>>>(agg, conv_b, Wi, Wh, bi, bh, hidden);
    }

    // 4) mpnn = leaky_relu(hidden), in place
    lrelu_kernel<<<(N_NODES * H1D + 255) / 256, 256, 0, stream>>>(hidden, N_NODES * H1D);
    float* mpnn = hidden;
    float* s_b  = agg;   // reuse

    // 5) level embeddings (sum over P, then N-level normalize, ddof=1, eps 1e-8)
    level_sum_kernel<<<N_NODES, 128, 0, stream>>>(mpnn, b_idx, s_b);
    colstats_kernel<<<H1D, 256, 0, stream>>>(s_b, N_NODES, H1D, 1.f, 1e-8f, bM, bR);
    norm_inplace_kernel<<<(N_NODES * H1D + 255) / 256, 256, 0, stream>>>(s_b, bM, bR, N_NODES * H1D, H1D);
    level_sum_kernel<<<N_NODES, 128, 0, stream>>>(mpnn, t_idx, s_t);
    colstats_kernel<<<H1D, 256, 0, stream>>>(s_t, N_NODES, H1D, 1.f, 1e-8f, tM, tR);
    norm_inplace_kernel<<<(N_NODES * H1D + 255) / 256, 256, 0, stream>>>(s_t, tM, tR, N_NODES * H1D, H1D);

    // 6) legal-action gathers + their normalization stats (ddof=0, eps 1e-6)
    gather_kernel<<<(A_ACT * H1D + 255) / 256, 256, 0, stream>>>(s_b, legal, gb, A_ACT * H1D, H1D);
    colstats_kernel<<<H1D, 256, 0, stream>>>(gb, A_ACT, H1D, 0.f, 1e-6f, gbm, gbr);
    gather_kernel<<<(A_ACT * H1D + 255) / 256, 256, 0, stream>>>(s_t, legal, gt, A_ACT * H1D, H1D);
    colstats_kernel<<<H1D, 256, 0, stream>>>(gt, A_ACT, H1D, 0.f, 1e-6f, gtm, gtr);
    gather_kernel<<<(A_ACT * FDIM + 255) / 256, 256, 0, stream>>>(feat, legal, lasg, A_ACT * FDIM, FDIM);
    colstats_kernel<<<FDIM, 256, 0, stream>>>(lasg, A_ACT, FDIM, 0.f, 1e-6f, lm, lr);

    // 7) rep assembly + FFNN head
    build_rep_kernel<<<A_ACT, 512, 0, stream>>>(lasg, lm, lr, W1, b1, mpnn, legal,
                                                gb, gbm, gbr, gt, gtm, gtr, rep);
    head_kernel<<<A_ACT, 64, 0, stream>>>(rep, W2, b2, W3, b3, out);
}
